// AttentionFlow_25323127177413
// MI455X (gfx1250) — compile-verified, matrix operand order checked by reading, not run on hardware
//
#include <hip/hip_runtime.h>
#include <hip/hip_bf16.h>

typedef __attribute__((ext_vector_type(16))) _Float16 v16h;
typedef __attribute__((ext_vector_type(8)))  float    v8f;

#define NDIM 64

// Order-preserving float <-> signed-int key (for atomicMax-based segment max).
__device__ __forceinline__ int f32_key(float f) {
    int b = __float_as_int(f);
    return b >= 0 ? b : (b ^ 0x7fffffff);
}
__device__ __forceinline__ float key_f32(int k) {
    int b = k >= 0 ? k : (k ^ 0x7fffffff);
    return __int_as_float(b);
}

// ---------------------------------------------------------------------------
// Kernel 1: per-node preprocessing.
//   U[n] = { relu(h)*pw2 , -neg(h)*nw2 }   (128 x f16)  -> WMMA A-side
//   V[n] = { relu(h)     ,  neg(h)     }   (128 x f16)  -> WMMA B-side
//   a[n] = sum_d relu*pw0 - neg*nw0,  b[n] = sum_d relu*pw1 - neg*nw1
//   also zero-init keys / denom / out (node-indexed, so folded in here).
// One block (64 threads = 2 waves) per node.
// ---------------------------------------------------------------------------
__global__ __launch_bounds__(64)
void prep_kernel(const float* __restrict__ hidden,
                 const float* __restrict__ pw,
                 const float* __restrict__ nw,
                 _Float16* __restrict__ U, _Float16* __restrict__ V,
                 float* __restrict__ a, float* __restrict__ b,
                 int* __restrict__ keys, float* __restrict__ denom,
                 float* __restrict__ out, int n_nodes) {
    int n = blockIdx.x;
    if (n >= n_nodes) return;
    int d = threadIdx.x;                       // 0..63
    float h   = hidden[(long)n * NDIM + d];
    float pos = fmaxf(h, 0.0f);
    float neg = -fminf(h, 0.0f);

    long base = (long)n * 2 * NDIM;
    U[base + d]        = (_Float16)( pos * pw[2 * NDIM + d]);
    U[base + NDIM + d] = (_Float16)(-neg * nw[2 * NDIM + d]);
    V[base + d]        = (_Float16)pos;
    V[base + NDIM + d] = (_Float16)neg;

    float av = pos * pw[d]        - neg * nw[d];
    float bv = pos * pw[NDIM + d] - neg * nw[NDIM + d];
    #pragma unroll
    for (int off = 16; off > 0; off >>= 1) {
        av += __shfl_down(av, off, 32);
        bv += __shfl_down(bv, off, 32);
    }
    __shared__ float sa[2], sb[2];
    int wid = threadIdx.x >> 5, lid = threadIdx.x & 31;
    if (lid == 0) { sa[wid] = av; sb[wid] = bv; }
    __syncthreads();
    if (threadIdx.x == 0) {
        a[n]    = sa[0] + sa[1];
        b[n]    = sb[0] + sb[1];
        keys[n] = (int)0x80000000;             // -inf key
        denom[n] = 0.0f;
    }
    out[(long)n * NDIM + d] = 0.0f;
}

// ---------------------------------------------------------------------------
// Kernel 2: per-edge logits via WMMA.
// One wave32 handles 16 edges: A(16x128 f16) rows = U[vi], B(128x16 f16)
// cols = V[vj]; 4x v_wmma_f32_16x16x32_f16 accumulate K=128; C diagonal is
// the cross term. logit = a[vi] + b[vj] + cross; atomicMax segment max on vi.
// Fragment layout used here (contiguous k, not the ISA table order): lane l supplies a contiguous 16-half
// run at K-offset 32*chunk + 16*(l>>4), row/col = l&15 — for both A and B.
// ---------------------------------------------------------------------------
__global__ __launch_bounds__(256)
void score_kernel(const int* __restrict__ edges,
                  const _Float16* __restrict__ U,
                  const _Float16* __restrict__ V,
                  const float* __restrict__ a,
                  const float* __restrict__ b,
                  float* __restrict__ logits,
                  int* __restrict__ keys,
                  int n_edges) {
    const int lane  = threadIdx.x & 31;
    const int wave  = blockIdx.x * (blockDim.x >> 5) + (threadIdx.x >> 5);
    const int eBase = wave * 16;
    if (eBase >= n_edges) return;              // wave-uniform exit: EXEC stays full

    const int row = lane & 15;
    const int hi  = lane >> 4;
    const int e   = min(eBase + row, n_edges - 1);   // clamp tail, keep EXEC all-1s
    const int vi  = edges[(long)e * 8 + 1];
    const int vj  = edges[(long)e * 8 + 2];

    const _Float16* uPtr = U + (long)vi * 128 + 16 * hi;
    const _Float16* vPtr = V + (long)vj * 128 + 16 * hi;

    v8f c = {};
    #pragma unroll
    for (int chunk = 0; chunk < 4; ++chunk) {
        v16h av = *(const v16h*)(uPtr + 32 * chunk);
        v16h bv = *(const v16h*)(vPtr + 32 * chunk);
        c = __builtin_amdgcn_wmma_f32_16x16x32_f16(
                /*neg_a=*/false, av, /*neg_b=*/false, bv,
                /*c_mod=*/(short)0, c, /*reuse_a=*/false, /*reuse_b=*/false);
    }

    // Diagonal ownership (32-bit C 16x16 layout): lanes 0..7 hold (m=r,n=lane)
    // at VGPR r=lane; lanes 24..31 hold (m=8+r, n=lane-16) at r=lane-24.
    float carr[8];
    #pragma unroll
    for (int i = 0; i < 8; ++i) carr[i] = c[i];

    int m = -1, r = 0;
    if (lane < 8)        { m = lane;      r = lane;      }
    else if (lane >= 24) { m = lane - 16; r = lane - 24; }

    if (m >= 0) {
        int em = eBase + m;
        if (em < n_edges) {
            int evi = edges[(long)em * 8 + 1];
            int evj = edges[(long)em * 8 + 2];
            float lg = a[evi] + b[evj] + carr[r];
            logits[em] = lg;
            atomicMax(&keys[evi], f32_key(lg));
        }
    }
}

// ---------------------------------------------------------------------------
// Kernel 3: ex = exp(logit - segmax[vi]); denom[vi] += ex.
// ---------------------------------------------------------------------------
__global__ __launch_bounds__(256)
void expsum_kernel(const int* __restrict__ edges,
                   const float* __restrict__ logits,
                   const int* __restrict__ keys,
                   float* __restrict__ exbuf,
                   float* __restrict__ denom,
                   int n_edges) {
    int e = blockIdx.x * blockDim.x + threadIdx.x;
    if (e >= n_edges) return;
    int vi   = edges[(long)e * 8 + 1];
    float ex = __expf(logits[e] - key_f32(keys[vi]));
    exbuf[e] = ex;
    atomicAdd(&denom[vi], ex);
}

// ---------------------------------------------------------------------------
// Kernel 4: out[vj] += (ex/denom[vi]) * hidden[vi]   (one thread per (e,d)).
// ---------------------------------------------------------------------------
__global__ __launch_bounds__(256)
void scatter_kernel(const int* __restrict__ edges,
                    const float* __restrict__ hidden,
                    const float* __restrict__ exbuf,
                    const float* __restrict__ denom,
                    float* __restrict__ out,
                    long n_work) {
    long idx = (long)blockIdx.x * blockDim.x + threadIdx.x;
    if (idx >= n_work) return;
    int e = (int)(idx >> 6);
    int d = (int)(idx & 63);
    int vi = edges[(long)e * 8 + 1];
    int vj = edges[(long)e * 8 + 2];
    float attn = exbuf[e] / denom[vi];
    atomicAdd(&out[(long)vj * NDIM + d], attn * hidden[(long)vi * NDIM + d]);
}

// ---------------------------------------------------------------------------
extern "C" void kernel_launch(void* const* d_in, const int* in_sizes, int n_in,
                              void* d_out, int out_size, void* d_ws, size_t ws_size,
                              hipStream_t stream) {
    const float* hidden = (const float*)d_in[0];   // (1, N, 64) f32
    const float* pw     = (const float*)d_in[1];   // (3, 64)    f32
    const float* nw     = (const float*)d_in[2];   // (3, 64)    f32
    const int*   edges  = (const int*)d_in[3];     // (E, 8)     i32
    float*       out    = (float*)d_out;           // (1, N, 64) f32

    const int n_nodes = in_sizes[0] / NDIM;
    const int n_edges = in_sizes[3] / 8;

    // Workspace carve-out (256B aligned slices).
    char* ws = (char*)d_ws;
    size_t off = 0;
    auto carve = [&](size_t bytes) -> void* {
        void* p = ws + off;
        off += (bytes + 255) & ~(size_t)255;
        return p;
    };
    _Float16* U      = (_Float16*)carve((size_t)n_nodes * 128 * sizeof(_Float16));
    _Float16* V      = (_Float16*)carve((size_t)n_nodes * 128 * sizeof(_Float16));
    float*    a      = (float*)carve((size_t)n_nodes * sizeof(float));
    float*    b      = (float*)carve((size_t)n_nodes * sizeof(float));
    int*      keys   = (int*)carve((size_t)n_nodes * sizeof(int));
    float*    denom  = (float*)carve((size_t)n_nodes * sizeof(float));
    float*    logits = (float*)carve((size_t)n_edges * sizeof(float));
    float*    exbuf  = (float*)carve((size_t)n_edges * sizeof(float));

    // 1) node prep + zero-init
    prep_kernel<<<n_nodes, 64, 0, stream>>>(hidden, pw, nw, U, V, a, b,
                                            keys, denom, out, n_nodes);

    // 2) edge logits via WMMA (16 edges / wave, 8 waves / block)
    int waves  = (n_edges + 15) / 16;
    int blocks = (waves + 7) / 8;
    score_kernel<<<blocks, 256, 0, stream>>>(edges, U, V, a, b, logits, keys, n_edges);

    // 3) exp + segment-sum denominator
    expsum_kernel<<<(n_edges + 255) / 256, 256, 0, stream>>>(edges, logits, keys,
                                                             exbuf, denom, n_edges);

    // 4) weighted scatter into out[vj]
    long n_work = (long)n_edges * NDIM;
    long sblocks = (n_work + 255) / 256;
    scatter_kernel<<<(unsigned)sblocks, 256, 0, stream>>>(edges, hidden, exbuf,
                                                          denom, out, n_work);
}
